// CrossAttentionBlock_31138512896545
// MI455X (gfx1250) — matrix operand order repaired and checked by reading, not compiled, not run on hardware
//
#include <hip/hip_runtime.h>

// ---------------------------------------------------------------------------
// MI455X (gfx1250, wave32) implementation of the PointTransformer cross-
// attention block. GEMMs use v_wmma_f32_16x16x32_f16 with f32 accumulation;
// BatchNorm (training mode) is done with two-level reductions (registers ->
// f32 global atomics) and the affine is fused into GEMM A-fragment loads.
// ---------------------------------------------------------------------------

typedef __attribute__((ext_vector_type(16))) _Float16 v16h;
typedef __attribute__((ext_vector_type(8)))  _Float16 v8h;
typedef __attribute__((ext_vector_type(8)))  float    v8f;

#define K_N1 16384
#define K_N2 16384
#define K_C  256
#define K_NS 16
#define K_CS 32      // C // S
#define K_EPS 1e-5f

// ---------------------------------------------------------------------------
// Utility kernels
// ---------------------------------------------------------------------------
__global__ __launch_bounds__(256) void zero_f32(float* __restrict__ p, long n) {
  for (long i = (long)blockIdx.x * 256 + threadIdx.x; i < n; i += (long)gridDim.x * 256)
    p[i] = 0.0f;
}

__global__ __launch_bounds__(256) void cvt_f16(const float* __restrict__ in,
                                               _Float16* __restrict__ out, long n) {
  for (long i = (long)blockIdx.x * 256 + threadIdx.x; i < n; i += (long)gridDim.x * 256)
    out[i] = (_Float16)in[i];
}

__global__ __launch_bounds__(256) void copy_f32(const float* __restrict__ in,
                                                float* __restrict__ out, long n) {
  for (long i = (long)blockIdx.x * 256 + threadIdx.x; i < n; i += (long)gridDim.x * 256)
    out[i] = in[i];
}

__global__ void copy_i32_1(const int* __restrict__ in, int* __restrict__ out) {
  out[0] = in[0];
}

// ---------------------------------------------------------------------------
// Per-channel sum / sum-of-squares (training-mode BN stats).
// Channel is the innermost axis; thread owns one channel, strides over rows,
// then does exactly two f32 global atomics.
// ---------------------------------------------------------------------------
template <typename T>
__global__ __launch_bounds__(256)
void stats_kernel(const T* __restrict__ x, float* __restrict__ sum,
                  float* __restrict__ sumsq, long rows, int ch) {
  int groups = 256 / ch;
  int g = threadIdx.x / ch;
  int c = threadIdx.x - g * ch;
  if (g >= groups) return;
  float s = 0.0f, q = 0.0f;
  for (long r = (long)blockIdx.x * groups + g; r < rows; r += (long)gridDim.x * groups) {
    float v = (float)x[r * ch + c];
    s += v;
    q += v * v;
  }
  atomicAdd(&sum[c], s);
  atomicAdd(&sumsq[c], q);
}

// ---------------------------------------------------------------------------
// WMMA GEMM:  OUT[M,N] = pre(A16[M,K]) @ W16[N,K]^T (+ bias)
//   pre() = optional per-K-channel BN affine + ReLU (stats folded in LDS).
// Wave computes a 16 x (16*NT) tile; 8 waves per block.
//   N=256 -> NT=4 (4 waves along N, 2 along M, block = 32 rows)
//   N=32  -> NT=2 (1 wave  along N, 8 along M, block = 128 rows)
// ---------------------------------------------------------------------------
template <int NT, bool PRE_BN, bool BIAS, typename OutT>
__global__ void __launch_bounds__(256)
gemm_wmma(const _Float16* __restrict__ A, const _Float16* __restrict__ W,
          const float* __restrict__ bias,
          const float* __restrict__ sum, const float* __restrict__ sumsq,
          const float* __restrict__ gamma, const float* __restrict__ beta,
          float invCnt, OutT* __restrict__ out, int M, int N, int K) {
  __shared__ float sA[256], sB[256];
  if (PRE_BN) {
    for (int c = threadIdx.x; c < K; c += 256) {
      float m = sum[c] * invCnt;
      float var = sumsq[c] * invCnt - m * m;
      float a = gamma[c] * rsqrtf(var + K_EPS);
      sA[c] = a;
      sB[c] = beta[c] - m * a;
    }
    __syncthreads();
  }

  const int lane   = threadIdx.x & 31;
  const int wave   = threadIdx.x >> 5;
  const int wavesN = N / (16 * NT);
  const int wavesM = 8 / wavesN;
  const int wn = wave % wavesN;
  const int wm = wave / wavesN;
  const int rowBase = blockIdx.x * (wavesM * 16) + wm * 16;
  const int colBase = wn * (16 * NT);
  const int lhalf = (lane >= 16) ? 1 : 0;
  const int lmod  = lane & 15;
  const int row   = rowBase + lmod;

  v8f acc[NT] = {};

  for (int kb = 0; kb < K; kb += 32) {
    // --- A fragment: lane<16 holds K {kb..kb+7, kb+16..kb+23}, lane>=16 +8 ---
    const _Float16* ap = A + (size_t)row * K + kb + lhalf * 8;
    v8h lo = *(const v8h*)(ap);
    v8h hi = *(const v8h*)(ap + 16);
    if (PRE_BN) {
#pragma unroll
      for (int e = 0; e < 8; e++) {
        int k0 = kb + lhalf * 8 + e;
        float f0 = (float)lo[e] * sA[k0] + sB[k0];
        lo[e] = (_Float16)fmaxf(f0, 0.0f);
        int k1 = k0 + 16;
        float f1 = (float)hi[e] * sA[k1] + sB[k1];
        hi[e] = (_Float16)fmaxf(f1, 0.0f);
      }
    }
    v16h a;
#pragma unroll
    for (int e = 0; e < 8; e++) { a[e] = lo[e]; a[8 + e] = hi[e]; }

    // --- B fragments: lane holds column n=lmod, same K split as A fragment
#pragma unroll
    for (int t = 0; t < NT; t++) {
      const _Float16* wp = W + (size_t)(colBase + t * 16 + lmod) * K + kb + lhalf * 8;
      v8h blo = *(const v8h*)(wp);
      v8h bhi = *(const v8h*)(wp + 16);
      v16h b;
#pragma unroll
      for (int e = 0; e < 8; e++) { b[e] = blo[e]; b[8 + e] = bhi[e]; }
      acc[t] = __builtin_amdgcn_wmma_f32_16x16x32_f16(
          false, a, false, b, (short)0, acc[t], false, false);
    }
  }

  // --- store: lane<16: VGPR r -> M=r ; lane>=16: M=r+8 ---
#pragma unroll
  for (int t = 0; t < NT; t++) {
    int n = colBase + t * 16 + lmod;
    float bv = BIAS ? bias[n] : 0.0f;
#pragma unroll
    for (int r = 0; r < 8; r++) {
      int m = rowBase + r + lhalf * 8;
      out[(size_t)m * N + n] = (OutT)(acc[t][r] + bv);
    }
  }
}

// ---------------------------------------------------------------------------
// BN (fresh stats) + ReLU apply; optional f32 output + f16 output (ch == 256)
// ---------------------------------------------------------------------------
__global__ __launch_bounds__(256)
void bn_relu_apply(const float* __restrict__ h, const float* __restrict__ sum,
                   const float* __restrict__ sumsq, const float* __restrict__ gamma,
                   const float* __restrict__ beta, float invCnt,
                   float* __restrict__ out32, _Float16* __restrict__ out16,
                   long total, int ch) {
  __shared__ float sa[256], sb[256];
  for (int c = threadIdx.x; c < ch; c += 256) {
    float m = sum[c] * invCnt;
    float v = sumsq[c] * invCnt - m * m;
    float a = gamma[c] * rsqrtf(v + K_EPS);
    sa[c] = a;
    sb[c] = beta[c] - m * a;
  }
  __syncthreads();
  for (long i = (long)blockIdx.x * 256 + threadIdx.x; i < total; i += (long)gridDim.x * 256) {
    int c = (int)(i & (ch - 1));
    float x = fmaxf(h[i] * sa[c] + sb[c], 0.0f);
    if (out32) out32[i] = x;
    out16[i] = (_Float16)x;
  }
}

// ---------------------------------------------------------------------------
// KNN: one thread per query point, p2 staged in LDS tiles of 256, top-16 kept
// in registers (static-indexed argmax replace -> no scratch spills).
// Downstream ops (softmax + sum over NS) are permutation invariant, so no
// final sort is required.
// ---------------------------------------------------------------------------
__global__ __launch_bounds__(256)
void knn_kernel(const float* __restrict__ p1, const float* __restrict__ p2,
                int* __restrict__ idx, int n1, int n2) {
  __shared__ float sx[256], sy[256], sz[256];
  int qi = blockIdx.x * 256 + threadIdx.x;
  float qx = 0.f, qy = 0.f, qz = 0.f;
  if (qi < n1) { qx = p1[qi * 3]; qy = p1[qi * 3 + 1]; qz = p1[qi * 3 + 2]; }

  float dbest[K_NS];
  int   ibest[K_NS];
#pragma unroll
  for (int j = 0; j < K_NS; j++) { dbest[j] = 3.4e38f; ibest[j] = 0; }
  float worst = 3.4e38f;

  for (int base = 0; base < n2; base += 256) {
    __syncthreads();
    int t = threadIdx.x;
    sx[t] = p2[(base + t) * 3];
    sy[t] = p2[(base + t) * 3 + 1];
    sz[t] = p2[(base + t) * 3 + 2];
    __syncthreads();
    for (int j = 0; j < 256; j++) {
      float dx = sx[j] - qx, dy = sy[j] - qy, dz = sz[j] - qz;
      float d = dx * dx + dy * dy + dz * dz;
      if (d < worst) {
        int wpos = 0;
        float wv = dbest[0];
#pragma unroll
        for (int u = 1; u < K_NS; u++)
          if (dbest[u] > wv) { wv = dbest[u]; wpos = u; }
#pragma unroll
        for (int u = 0; u < K_NS; u++)
          if (u == wpos) { dbest[u] = d; ibest[u] = base + j; }
        float nw = dbest[0];
#pragma unroll
        for (int u = 1; u < K_NS; u++) nw = fmaxf(nw, dbest[u]);
        worst = nw;
      }
    }
  }
  if (qi < n1) {
#pragma unroll
    for (int j = 0; j < K_NS; j++) idx[qi * K_NS + j] = ibest[j];
  }
}

// ---------------------------------------------------------------------------
// linear_p stage 1: pr3[(n,s),:] = (p2[idx]-p1) @ Wp1^T + bp1   (3x3, tiny)
// ---------------------------------------------------------------------------
__global__ __launch_bounds__(256)
void pr3_kernel(const float* __restrict__ p1, const float* __restrict__ p2,
                const int* __restrict__ idx, const float* __restrict__ Wp1,
                const float* __restrict__ bp1, float* __restrict__ pr3, long total) {
  long t = (long)blockIdx.x * 256 + threadIdx.x;
  if (t >= total) return;
  long n = t >> 4;  // / NS
  int  i2 = idx[t];
  float r0 = p2[i2 * 3]     - p1[n * 3];
  float r1 = p2[i2 * 3 + 1] - p1[n * 3 + 1];
  float r2 = p2[i2 * 3 + 2] - p1[n * 3 + 2];
#pragma unroll
  for (int j = 0; j < 3; j++)
    pr3[t * 3 + j] = Wp1[j * 3] * r0 + Wp1[j * 3 + 1] * r1 + Wp1[j * 3 + 2] * r2 + bp1[j];
}

// ---------------------------------------------------------------------------
// BN(3)+ReLU on pr3, linear (3->C), then  w = k[idx] - q + p_r.
// One block per (n,s); thread = channel c.  Stores p_r and w as f16.
// ---------------------------------------------------------------------------
__global__ __launch_bounds__(256)
void prw_kernel(const float* __restrict__ pr3, const float* __restrict__ sum3,
                const float* __restrict__ sumsq3, const float* __restrict__ gp,
                const float* __restrict__ bp, const float* __restrict__ Wp2,
                const float* __restrict__ bp2, const float* __restrict__ q,
                const float* __restrict__ k, const int* __restrict__ idx,
                _Float16* __restrict__ pr16, _Float16* __restrict__ w16,
                float invCnt) {
  long b = blockIdx.x;        // (n, s)
  long n = b >> 4;
  int  c = threadIdx.x;
  int  i2 = idx[b];

  float r[3];
#pragma unroll
  for (int j = 0; j < 3; j++) {
    float m  = sum3[j] * invCnt;
    float va = sumsq3[j] * invCnt - m * m;
    float a  = gp[j] * rsqrtf(va + K_EPS);
    float bb = bp[j] - m * a;
    r[j] = fmaxf(pr3[b * 3 + j] * a + bb, 0.0f);
  }
  float pr = Wp2[c * 3] * r[0] + Wp2[c * 3 + 1] * r[1] + Wp2[c * 3 + 2] * r[2] + bp2[c];
  float w  = k[(size_t)i2 * K_C + c] - q[n * K_C + c] + pr;
  pr16[b * K_C + c] = (_Float16)pr;
  w16[b * K_C + c]  = (_Float16)w;
}

// ---------------------------------------------------------------------------
// Softmax over NS for each (n, c32); the 16 values stay in registers.
// ---------------------------------------------------------------------------
__global__ __launch_bounds__(256)
void softmax_kernel(const float* __restrict__ in, float* __restrict__ out, long total) {
  long t = (long)blockIdx.x * 256 + threadIdx.x;  // (n, c32)
  if (t >= total) return;
  long n = t >> 5;
  int  c = (int)(t & 31);
  const float* p = in + n * K_NS * K_CS + c;
  float mx = -3.4e38f;
  float e[K_NS];
#pragma unroll
  for (int s = 0; s < K_NS; s++) mx = fmaxf(mx, p[s * K_CS]);
  float sum = 0.0f;
#pragma unroll
  for (int s = 0; s < K_NS; s++) { e[s] = __expf(p[s * K_CS] - mx); sum += e[s]; }
  float inv = 1.0f / sum;
  float* o = out + n * K_NS * K_CS + c;
#pragma unroll
  for (int s = 0; s < K_NS; s++) o[s * K_CS] = e[s] * inv;
}

// ---------------------------------------------------------------------------
// Aggregation: xout[n,c] = sum_s (v[idx[n,s],c] + p_r[n,s,c]) * attn[n,s,c%32]
// ---------------------------------------------------------------------------
__global__ __launch_bounds__(256)
void aggregate_kernel(const float* __restrict__ v, const _Float16* __restrict__ pr16,
                      const float* __restrict__ attn, const int* __restrict__ idx,
                      float* __restrict__ xout, long total) {
  long t = (long)blockIdx.x * 256 + threadIdx.x;  // (n, c)
  if (t >= total) return;
  long n = t >> 8;
  int  c = (int)(t & 255);
  int  c32 = c & 31;
  float acc = 0.0f;
#pragma unroll
  for (int s = 0; s < K_NS; s++) {
    long ns = n * K_NS + s;
    int  i2 = idx[ns];
    acc += (v[(size_t)i2 * K_C + c] + (float)pr16[ns * K_C + c]) * attn[ns * K_CS + c32];
  }
  xout[t] = acc;
}

// ---------------------------------------------------------------------------
// Final: y = bn3(h3); out = relu(y + x1h)
// ---------------------------------------------------------------------------
__global__ __launch_bounds__(256)
void final_kernel(const float* __restrict__ h3, const float* __restrict__ sum,
                  const float* __restrict__ sumsq, const float* __restrict__ g3,
                  const float* __restrict__ b3, const float* __restrict__ x1h,
                  float invCnt, float* __restrict__ outy, long total) {
  __shared__ float sa[256], sb[256];
  for (int c = threadIdx.x; c < K_C; c += 256) {
    float m = sum[c] * invCnt;
    float v = sumsq[c] * invCnt - m * m;
    float a = g3[c] * rsqrtf(v + K_EPS);
    sa[c] = a;
    sb[c] = b3[c] - m * a;
  }
  __syncthreads();
  for (long i = (long)blockIdx.x * 256 + threadIdx.x; i < total; i += (long)gridDim.x * 256) {
    int c = (int)(i & 255);
    float y = h3[i] * sa[c] + sb[c];
    outy[i] = fmaxf(y + x1h[i], 0.0f);
  }
}

// ---------------------------------------------------------------------------
// Host-side orchestration
// ---------------------------------------------------------------------------
extern "C" void kernel_launch(void* const* d_in, const int* in_sizes, int n_in,
                              void* d_out, int out_size, void* d_ws, size_t ws_size,
                              hipStream_t stream) {
  (void)in_sizes; (void)n_in; (void)out_size; (void)ws_size;

  const long N1 = K_N1, N2 = K_N2, C = K_C, NS = K_NS, CS = K_CS;

  // --- inputs (setup_inputs order) ---
  const float* p1  = (const float*)d_in[0];
  const float* x1  = (const float*)d_in[1];
  const int*   o1  = (const int*)  d_in[2];
  const float* p2  = (const float*)d_in[3];
  const float* x2  = (const float*)d_in[4];
  const float* W1  = (const float*)d_in[6];
  const float* g1  = (const float*)d_in[7];
  const float* b1  = (const float*)d_in[8];
  const float* W2  = (const float*)d_in[9];
  const float* g2  = (const float*)d_in[10];
  const float* b2  = (const float*)d_in[11];
  const float* Wq  = (const float*)d_in[12];
  const float* bq  = (const float*)d_in[13];
  const float* Wk  = (const float*)d_in[14];
  const float* bk  = (const float*)d_in[15];
  const float* Wv  = (const float*)d_in[16];
  const float* bv  = (const float*)d_in[17];
  const float* Wp1 = (const float*)d_in[18];
  const float* bp1 = (const float*)d_in[19];
  const float* gp  = (const float*)d_in[20];
  const float* bp  = (const float*)d_in[21];
  const float* Wp2 = (const float*)d_in[22];
  const float* bp2 = (const float*)d_in[23];
  const float* gw1 = (const float*)d_in[24];
  const float* bw1 = (const float*)d_in[25];
  const float* Ww1 = (const float*)d_in[26];
  const float* bww1= (const float*)d_in[27];
  const float* gw2 = (const float*)d_in[28];
  const float* bw2 = (const float*)d_in[29];
  const float* Ww2 = (const float*)d_in[30];
  const float* bww2= (const float*)d_in[31];
  const float* W3  = (const float*)d_in[32];
  const float* g3  = (const float*)d_in[33];
  const float* b3  = (const float*)d_in[34];

  // --- workspace bump allocator ---
  char* base = (char*)d_ws;
  size_t off = 0;
  auto alloc = [&](size_t bytes) -> void* {
    void* p = base + off;
    off = (off + bytes + 255) & ~(size_t)255;
    return p;
  };

  _Float16* W1h  = (_Float16*)alloc(C * C * 2);
  _Float16* W2h  = (_Float16*)alloc(C * C * 2);
  _Float16* Wqh  = (_Float16*)alloc(C * C * 2);
  _Float16* Wkh  = (_Float16*)alloc(C * C * 2);
  _Float16* Wvh  = (_Float16*)alloc(C * C * 2);
  _Float16* W3h  = (_Float16*)alloc(C * C * 2);
  _Float16* Ww1h = (_Float16*)alloc(CS * C * 2);
  _Float16* Ww2h = (_Float16*)alloc(CS * CS * 2);
  _Float16* x1_16 = (_Float16*)alloc(N1 * C * 2);
  _Float16* x2_16 = (_Float16*)alloc(N2 * C * 2);
  float* h1    = (float*)alloc(N1 * C * 4);
  float* h2    = (float*)alloc(N2 * C * 4);
  float* x1h32 = (float*)alloc(N1 * C * 4);
  _Float16* x1h16 = (_Float16*)alloc(N1 * C * 2);
  _Float16* x2h16 = (_Float16*)alloc(N2 * C * 2);
  float* qb = (float*)alloc(N1 * C * 4);
  float* kb_ = (float*)alloc(N2 * C * 4);
  float* vb = (float*)alloc(N2 * C * 4);
  int*   idx = (int*)alloc(N1 * NS * 4);
  float* pr3 = (float*)alloc(N1 * NS * 3 * 4);
  _Float16* pr16 = (_Float16*)alloc(N1 * NS * C * 2);
  _Float16* w16  = (_Float16*)alloc(N1 * NS * C * 2);
  _Float16* w1_16 = (_Float16*)alloc(N1 * NS * CS * 2);
  float* attnpre = (float*)alloc(N1 * NS * CS * 4);
  float* attn    = (float*)alloc(N1 * NS * CS * 4);
  float* xout = (float*)alloc(N1 * C * 4);
  _Float16* y1_16 = (_Float16*)alloc(N1 * C * 2);
  float* h3 = (float*)alloc(N1 * C * 4);
  float* stats = (float*)alloc(16 * 256 * 4);   // 14 slots of 256 f32 used

  float* s_bn1  = stats + 0 * 256;  float* q_bn1  = stats + 1 * 256;
  float* s_bn2  = stats + 2 * 256;  float* q_bn2  = stats + 3 * 256;
  float* s_p    = stats + 4 * 256;  float* q_p    = stats + 5 * 256;
  float* s_w1   = stats + 6 * 256;  float* q_w1   = stats + 7 * 256;
  float* s_w2   = stats + 8 * 256;  float* q_w2   = stats + 9 * 256;
  float* s_x    = stats + 10 * 256; float* q_x    = stats + 11 * 256;
  float* s_h3   = stats + 12 * 256; float* q_h3   = stats + 13 * 256;

  // --- phase 0: zero stats, convert weights & inputs to f16 ---
  zero_f32<<<4, 256, 0, stream>>>(stats, 16 * 256);
  cvt_f16<<<256, 256, 0, stream>>>(W1, W1h, C * C);
  cvt_f16<<<256, 256, 0, stream>>>(W2, W2h, C * C);
  cvt_f16<<<256, 256, 0, stream>>>(Wq, Wqh, C * C);
  cvt_f16<<<256, 256, 0, stream>>>(Wk, Wkh, C * C);
  cvt_f16<<<256, 256, 0, stream>>>(Wv, Wvh, C * C);
  cvt_f16<<<256, 256, 0, stream>>>(W3, W3h, C * C);
  cvt_f16<<<32, 256, 0, stream>>>(Ww1, Ww1h, CS * C);
  cvt_f16<<<4, 256, 0, stream>>>(Ww2, Ww2h, CS * CS);
  cvt_f16<<<1024, 256, 0, stream>>>(x1, x1_16, N1 * C);
  cvt_f16<<<1024, 256, 0, stream>>>(x2, x2_16, N2 * C);

  // --- phase 1: h1 = x1@W1^T, h2 = x2@W2^T, BN+ReLU ---
  gemm_wmma<4, false, false, float><<<N1 / 32, 256, 0, stream>>>(
      x1_16, W1h, nullptr, nullptr, nullptr, nullptr, nullptr, 0.f, h1, (int)N1, (int)C, (int)C);
  gemm_wmma<4, false, false, float><<<N2 / 32, 256, 0, stream>>>(
      x2_16, W2h, nullptr, nullptr, nullptr, nullptr, nullptr, 0.f, h2, (int)N2, (int)C, (int)C);
  stats_kernel<float><<<128, 256, 0, stream>>>(h1, s_bn1, q_bn1, N1, (int)C);
  stats_kernel<float><<<128, 256, 0, stream>>>(h2, s_bn2, q_bn2, N2, (int)C);
  bn_relu_apply<<<4096, 256, 0, stream>>>(h1, s_bn1, q_bn1, g1, b1, 1.0f / N1,
                                          x1h32, x1h16, N1 * C, (int)C);
  bn_relu_apply<<<4096, 256, 0, stream>>>(h2, s_bn2, q_bn2, g2, b2, 1.0f / N2,
                                          (float*)nullptr, x2h16, N2 * C, (int)C);

  // --- phase 2: q, k, v ---
  gemm_wmma<4, false, true, float><<<N1 / 32, 256, 0, stream>>>(
      x1h16, Wqh, bq, nullptr, nullptr, nullptr, nullptr, 0.f, qb, (int)N1, (int)C, (int)C);
  gemm_wmma<4, false, true, float><<<N2 / 32, 256, 0, stream>>>(
      x2h16, Wkh, bk, nullptr, nullptr, nullptr, nullptr, 0.f, kb_, (int)N2, (int)C, (int)C);
  gemm_wmma<4, false, true, float><<<N2 / 32, 256, 0, stream>>>(
      x2h16, Wvh, bv, nullptr, nullptr, nullptr, nullptr, 0.f, vb, (int)N2, (int)C, (int)C);

  // --- phase 3: KNN ---
  knn_kernel<<<N1 / 256, 256, 0, stream>>>(p1, p2, idx, (int)N1, (int)N2);

  // --- phase 4: linear_p + w = k[idx] - q + p_r ---
  pr3_kernel<<<(unsigned)((N1 * NS + 255) / 256), 256, 0, stream>>>(
      p1, p2, idx, Wp1, bp1, pr3, N1 * NS);
  stats_kernel<float><<<256, 256, 0, stream>>>(pr3, s_p, q_p, N1 * NS, 3);
  prw_kernel<<<(unsigned)(N1 * NS), 256, 0, stream>>>(
      pr3, s_p, q_p, gp, bp, Wp2, bp2, qb, kb_, idx, pr16, w16, 1.0f / (N1 * NS));

  // --- phase 5: linear_w (BN+ReLU fused into WMMA A-fragment loads) ---
  stats_kernel<_Float16><<<512, 256, 0, stream>>>(w16, s_w1, q_w1, N1 * NS, (int)C);
  gemm_wmma<2, true, true, _Float16><<<(unsigned)(N1 * NS / 128), 256, 0, stream>>>(
      w16, Ww1h, bww1, s_w1, q_w1, gw1, bw1, 1.0f / (N1 * NS),
      w1_16, (int)(N1 * NS), (int)CS, (int)C);
  stats_kernel<_Float16><<<512, 256, 0, stream>>>(w1_16, s_w2, q_w2, N1 * NS, (int)CS);
  gemm_wmma<2, true, true, float><<<(unsigned)(N1 * NS / 128), 256, 0, stream>>>(
      w1_16, Ww2h, bww2, s_w2, q_w2, gw2, bw2, 1.0f / (N1 * NS),
      attnpre, (int)(N1 * NS), (int)CS, (int)CS);
  softmax_kernel<<<(unsigned)((N1 * CS + 255) / 256), 256, 0, stream>>>(
      attnpre, attn, N1 * CS);

  // --- phase 6: aggregate, BN+ReLU, output projection, residual ---
  aggregate_kernel<<<(unsigned)(N1 * C / 256), 256, 0, stream>>>(
      vb, pr16, attn, idx, xout, N1 * C);
  stats_kernel<float><<<128, 256, 0, stream>>>(xout, s_x, q_x, N1, (int)C);
  bn_relu_apply<<<4096, 256, 0, stream>>>(xout, s_x, q_x, g2, b2, 1.0f / N1,
                                          (float*)nullptr, y1_16, N1 * C, (int)C);
  gemm_wmma<4, false, false, float><<<N1 / 32, 256, 0, stream>>>(
      y1_16, W3h, nullptr, nullptr, nullptr, nullptr, nullptr, 0.f, h3, (int)N1, (int)C, (int)C);
  stats_kernel<float><<<128, 256, 0, stream>>>(h3, s_h3, q_h3, N1, (int)C);

  // outputs: [p1 (N1*3)] [y (N1*C)] [o1 (1 int)]
  float* out_p1 = (float*)d_out;
  float* out_y  = out_p1 + N1 * 3;
  int*   out_o1 = (int*)((float*)d_out + N1 * 3 + N1 * C);

  final_kernel<<<4096, 256, 0, stream>>>(h3, s_h3, q_h3, g3, b3, x1h32,
                                         1.0f / N1, out_y, N1 * C);
  copy_f32<<<64, 256, 0, stream>>>(p1, out_p1, N1 * 3);
  copy_i32_1<<<1, 1, 0, stream>>>(o1, out_o1);
}
